// MambaBlock_54039278518897
// MI455X (gfx1250) — compile-verified, matrix operand order checked by reading, not run on hardware
//
#include <hip/hip_runtime.h>
#include <hip/hip_bf16.h>

#define D_MODEL  1024
#define D_STATE  16
#define D_CONV   4
#define DT_RANK  64
#define D_INNER  2048
#define BATCH    2
#define SEQLEN   2048
#define ROWS     (BATCH * SEQLEN)           // 4096
#define XDBL_W   (DT_RANK + 2 * D_STATE)    // 96

typedef __attribute__((ext_vector_type(16))) __bf16 v16bf;
typedef __attribute__((ext_vector_type(8)))  float  v8f;

// ---------------------------------------------------------------------------
// Load one 16x32 bf16 fragment (A or B operand) from an fp32 row-major source.
// `p` already points at element (row_of_this_lane, kb + 8*(lane>=16)).
// Per ISA 7.12.2 (16-bit A 16x32): lanes 0-15 hold K {0..7, 16..23},
// lanes 16-31 hold K {8..15, 24..31}; B (stored [N,K]) mirrors with lane = N.
// Both halves are 8 contiguous floats -> two float4 pairs.
// ---------------------------------------------------------------------------
__device__ inline v16bf load_frag_bf16(const float* __restrict__ p) {
    const float4* q = (const float4*)p;
    float4 a0 = q[0];   // K+0..3
    float4 a1 = q[1];   // K+4..7
    float4 a2 = q[4];   // K+16..19
    float4 a3 = q[5];   // K+20..23
    v16bf f;
    f[0]  = (__bf16)a0.x; f[1]  = (__bf16)a0.y; f[2]  = (__bf16)a0.z; f[3]  = (__bf16)a0.w;
    f[4]  = (__bf16)a1.x; f[5]  = (__bf16)a1.y; f[6]  = (__bf16)a1.z; f[7]  = (__bf16)a1.w;
    f[8]  = (__bf16)a2.x; f[9]  = (__bf16)a2.y; f[10] = (__bf16)a2.z; f[11] = (__bf16)a2.w;
    f[12] = (__bf16)a3.x; f[13] = (__bf16)a3.y; f[14] = (__bf16)a3.z; f[15] = (__bf16)a3.w;
    return f;
}

// ---------------------------------------------------------------------------
// Register-blocked wave-level GEMM: each wave owns a (16*MT) x (16*NT) macro
// tile of C = epi( A[M,K] @ W[N,K]^T ).  Per k-step: MT+NT fragment loads feed
// MT*NT WMMAs (32x64 tile -> 6 loads : 8 WMMAs, AI ~10.7 FLOP/B from L2).
// EPI 0: plain store.  EPI 1: += bias[n], softplus (for delta).
// ---------------------------------------------------------------------------
template <int MT, int NT, int EPI>
__global__ void gemm_wmma_bf16(const float* __restrict__ A, int lda,
                               const float* __restrict__ W, int ldw,
                               float* __restrict__ C, int ldc,
                               int M, int N, int K,
                               const float* __restrict__ bias) {
    const int wave   = (blockIdx.x * blockDim.x + threadIdx.x) >> 5;
    const int lane   = threadIdx.x & 31;
    const int nBlk   = N / (16 * NT);
    const int mBlk   = M / (16 * MT);
    if (wave >= mBlk * nBlk) return;
    const int tm = wave / nBlk;          // macro-tile row
    const int tn = wave % nBlk;          // macro-tile col

    const int half   = lane >> 4;        // 0 or 1
    const int kshift = half << 3;        // +8 K-offset for upper half-lanes
    const float* pa = A + (size_t)(tm * MT * 16 + (lane & 15)) * lda + kshift;
    const float* pw = W + (size_t)(tn * NT * 16 + (lane & 15)) * ldw + kshift;

    v8f acc[MT][NT];
#pragma unroll
    for (int i = 0; i < MT; ++i)
#pragma unroll
        for (int j = 0; j < NT; ++j) acc[i][j] = (v8f){};

    for (int kb = 0; kb < K; kb += 32) {
        v16bf af[MT], bfr[NT];
#pragma unroll
        for (int i = 0; i < MT; ++i)
            af[i] = load_frag_bf16(pa + (size_t)i * 16 * lda + kb);
#pragma unroll
        for (int j = 0; j < NT; ++j)
            bfr[j] = load_frag_bf16(pw + (size_t)j * 16 * ldw + kb);
#pragma unroll
        for (int i = 0; i < MT; ++i)
#pragma unroll
            for (int j = 0; j < NT; ++j)
                acc[i][j] = __builtin_amdgcn_wmma_f32_16x16x32_bf16(
                    false, af[i], false, bfr[j],
                    (short)0, acc[i][j], false, false);
    }

    // C/D layout: VGPR i -> M = 8*half + i, N = lane&15  (ISA 7.12.2)
#pragma unroll
    for (int j = 0; j < NT; ++j) {
        const int n = tn * NT * 16 + j * 16 + (lane & 15);
        const float b = (EPI == 1) ? bias[n] : 0.0f;
#pragma unroll
        for (int i = 0; i < MT; ++i) {
            const int mbase = tm * MT * 16 + i * 16 + (half << 3);
#pragma unroll
            for (int r = 0; r < 8; ++r) {
                float v = acc[i][j][r];
                if (EPI == 1) {
                    v += b;
                    v = (v > 20.0f) ? v : log1pf(__expf(v));   // softplus
                }
                C[(size_t)(mbase + r) * ldc + n] = v;
            }
        }
    }
}

// ---------------------------------------------------------------------------
// Causal depthwise conv1d (k=4) + bias + SiLU.  Reads x_ssm half of xz.
// ---------------------------------------------------------------------------
__global__ void conv_silu_kernel(const float* __restrict__ xz,
                                 const float* __restrict__ cw,
                                 const float* __restrict__ cb,
                                 float* __restrict__ out) {
    int idx = blockIdx.x * blockDim.x + threadIdx.x;
    if (idx >= BATCH * SEQLEN * D_INNER) return;
    int d = idx & (D_INNER - 1);
    int l = (idx / D_INNER) % SEQLEN;
    int b = idx / (D_INNER * SEQLEN);
    float s = cb[d];
#pragma unroll
    for (int j = 0; j < D_CONV; ++j) {
        int li = l + j - (D_CONV - 1);
        if (li >= 0)
            s += cw[d * D_CONV + j] *
                 xz[((size_t)(b * SEQLEN + li)) * (2 * D_INNER) + d];
    }
    out[idx] = s / (1.0f + __expf(-s));   // SiLU
}

// ---------------------------------------------------------------------------
// Selective scan. One lane per (b, d, n) state; 16 lanes per channel reduce
// y = sum_n h*C via shfl_xor inside the wave (wave32 -> 2 channels / wave).
// Epilogue: y = (y + u*D) * silu(z), written once per channel.
// ---------------------------------------------------------------------------
__global__ void scan_kernel(const float* __restrict__ u,      // [ROWS, D_INNER]
                            const float* __restrict__ delta,  // [ROWS, D_INNER]
                            const float* __restrict__ xdbl,   // [ROWS, 96]
                            const float* __restrict__ A_log,  // [D_INNER, 16]
                            const float* __restrict__ Dskip,  // [D_INNER]
                            const float* __restrict__ xz,     // [ROWS, 2*D_INNER] (z half)
                            float* __restrict__ ygate) {      // [ROWS, D_INNER]
    int tid = blockIdx.x * blockDim.x + threadIdx.x;   // 0 .. 65535
    int n = tid & 15;
    int c = tid >> 4;                  // channel 0..4095
    int d = c & (D_INNER - 1);
    int b = c >> 11;                   // D_INNER = 2048
    float Ad = -__expf(A_log[d * D_STATE + n]);
    float Dd = Dskip[d];
    float h = 0.0f;
    for (int t = 0; t < SEQLEN; ++t) {
        size_t row = (size_t)b * SEQLEN + t;
        float dl = delta[row * D_INNER + d];
        float ut = u[row * D_INNER + d];
        float Bn = xdbl[row * XDBL_W + DT_RANK + n];
        float Cn = xdbl[row * XDBL_W + DT_RANK + D_STATE + n];
        float dA = __expf(dl * Ad);
        h = dA * h + (dl * ut) * Bn;
        float y = h * Cn;
        y += __shfl_xor(y, 1, 32);
        y += __shfl_xor(y, 2, 32);
        y += __shfl_xor(y, 4, 32);
        y += __shfl_xor(y, 8, 32);
        if (n == 0) {
            float z  = xz[row * (2 * D_INNER) + D_INNER + d];
            float sz = z / (1.0f + __expf(-z));
            ygate[row * D_INNER + d] = (y + ut * Dd) * sz;
        }
    }
}

extern "C" void kernel_launch(void* const* d_in, const int* in_sizes, int n_in,
                              void* d_out, int out_size, void* d_ws, size_t ws_size,
                              hipStream_t stream) {
    const float* x         = (const float*)d_in[0];   // [B, L, 1024]
    const float* in_proj_w = (const float*)d_in[1];   // [4096, 1024]
    const float* conv_w    = (const float*)d_in[2];   // [2048, 1, 4]
    const float* conv_b    = (const float*)d_in[3];   // [2048]
    const float* x_proj_w  = (const float*)d_in[4];   // [96, 2048]
    const float* dt_proj_w = (const float*)d_in[5];   // [2048, 64]
    const float* dt_proj_b = (const float*)d_in[6];   // [2048]
    const float* A_log     = (const float*)d_in[7];   // [2048, 16]
    const float* D_skip    = (const float*)d_in[8];   // [2048]
    const float* out_proj_w= (const float*)d_in[9];   // [1024, 2048]
    float* out = (float*)d_out;                       // [B, L, 1024]

    // Workspace layout (floats)
    float* ws    = (float*)d_ws;
    float* xz    = ws;                                          // 4096*4096
    float* u     = xz    + (size_t)ROWS * 2 * D_INNER;          // 4096*2048
    float* xdbl  = u     + (size_t)ROWS * D_INNER;              // 4096*96
    float* delta = xdbl  + (size_t)ROWS * XDBL_W;               // 4096*2048
    float* ygate = delta + (size_t)ROWS * D_INNER;              // 4096*2048

    const dim3 blk(256);

    // 1) in_proj: xz = x @ in_proj_w^T      [4096 x 4096], K=1024   (32x64 tiles)
    {
        int waves = (ROWS / 32) * ((2 * D_INNER) / 64);         // 128*64 = 8192
        gemm_wmma_bf16<2, 4, 0><<<dim3(waves * 32 / 256), blk, 0, stream>>>(
            x, D_MODEL, in_proj_w, D_MODEL, xz, 2 * D_INNER,
            ROWS, 2 * D_INNER, D_MODEL, nullptr);
    }

    // 2) depthwise causal conv + SiLU -> u
    {
        int nthreads = BATCH * SEQLEN * D_INNER;
        conv_silu_kernel<<<dim3(nthreads / 256), blk, 0, stream>>>(
            xz, conv_w, conv_b, u);
    }

    // 3) x_proj: xdbl = u @ x_proj_w^T      [4096 x 96], K=2048     (32x32 tiles)
    {
        int waves = (ROWS / 32) * (XDBL_W / 32);                // 128*3 = 384
        gemm_wmma_bf16<2, 2, 0><<<dim3(waves * 32 / 256), blk, 0, stream>>>(
            u, D_INNER, x_proj_w, D_INNER, xdbl, XDBL_W,
            ROWS, XDBL_W, D_INNER, nullptr);
    }

    // 4) dt_proj + softplus: delta = softplus(xdbl[:, :64] @ dt_proj_w^T + b)
    {
        int waves = (ROWS / 32) * (D_INNER / 64);               // 128*32 = 4096
        gemm_wmma_bf16<2, 4, 1><<<dim3(waves * 32 / 256), blk, 0, stream>>>(
            xdbl, XDBL_W, dt_proj_w, DT_RANK, delta, D_INNER,
            ROWS, D_INNER, DT_RANK, dt_proj_b);
    }

    // 5) selective scan + D-skip + z-gating -> ygate
    {
        int nthreads = BATCH * D_INNER * D_STATE;               // 65536
        scan_kernel<<<dim3(nthreads / 256), blk, 0, stream>>>(
            u, delta, xdbl, A_log, D_skip, xz, ygate);
    }

    // 6) out_proj: out = ygate @ out_proj_w^T   [4096 x 1024], K=2048
    {
        int waves = (ROWS / 32) * (D_MODEL / 64);               // 128*16 = 2048
        gemm_wmma_bf16<2, 4, 0><<<dim3(waves * 32 / 256), blk, 0, stream>>>(
            ygate, D_INNER, out_proj_w, D_INNER, out, D_MODEL,
            ROWS, D_MODEL, D_INNER, nullptr);
    }
}
